// Loss_Multi_7438883356870
// MI455X (gfx1250) — matrix operand order repaired and checked by reading, not compiled, not run on hardware
//
#include <hip/hip_runtime.h>
#include <hip/hip_bf16.h>

// ---------------- problem constants (match reference) ----------------
#define N_BATCH     4096
#define NUM_CLASSES 751
#define FEAT_D      2048
#define MARGIN1     0.5f
#define BIG_DIST    1e12f

// ---------------- WMMA vector types (probe-confirmed) ----------------
typedef __attribute__((ext_vector_type(16))) __bf16 v16bf;
typedef __attribute__((ext_vector_type(8)))  __bf16 v8bf;
typedef __attribute__((ext_vector_type(8)))  float  v8f;

// fp32 -> bf16 hi/lo split (bf16x3 precision recovery)
struct BF16Pair { __bf16 hi; __bf16 lo; };
__device__ __forceinline__ BF16Pair split_bf16(float f) {
    BF16Pair p;
    __bf16 h = (__bf16)f;              // RNE conversion to bf16
    p.hi = h;
    p.lo = (__bf16)(f - (float)h);     // residual in bf16
    return p;
}

// assemble a 16-elem bf16 fragment from two 8-elem (b128) chunks `gap` apart
__device__ __forceinline__ v16bf load_frag(const __bf16* __restrict__ p, int gap) {
    v8bf a = *(const v8bf*)p;
    v8bf b = *(const v8bf*)(p + gap);
    v16bf r;
#pragma unroll
    for (int e = 0; e < 8; ++e) { r[e] = a[e]; r[8 + e] = b[e]; }
    return r;
}

// ---------------- kernel 0: init workspace ----------------
__global__ void init_kernel(unsigned* __restrict__ ap_bits,
                            unsigned* __restrict__ an_bits,
                            float* __restrict__ accum) {
    int i = blockIdx.x * blockDim.x + threadIdx.x;
    if (i < N_BATCH) {
        ap_bits[i] = 0u;                           // max accumulator, dist >= 0
        an_bits[i] = __float_as_uint(BIG_DIST);    // min accumulator
    }
    if (i == 0) accum[0] = 0.0f;
}

// ---------------- kernel 1: cross entropy (one block per row) ----------------
__global__ void xent_kernel(const float* __restrict__ logits,
                            const int*   __restrict__ targets,
                            float*       __restrict__ accum) {
    const int row = blockIdx.x;
    const float* lr = logits + (size_t)row * NUM_CLASSES;
    __shared__ float sm[8];
    const int lane = threadIdx.x & 31;
    const int wv   = threadIdx.x >> 5;

    float m = -3.4e38f;
    for (int j = threadIdx.x; j < NUM_CLASSES; j += 256)
        m = fmaxf(m, lr[j]);
    for (int s = 16; s; s >>= 1) m = fmaxf(m, __shfl_xor(m, s, 32));
    if (lane == 0) sm[wv] = m;
    __syncthreads();
    m = sm[0];
#pragma unroll
    for (int w = 1; w < 8; ++w) m = fmaxf(m, sm[w]);
    __syncthreads();

    float s = 0.0f;
    for (int j = threadIdx.x; j < NUM_CLASSES; j += 256)
        s += __expf(lr[j] - m);
    for (int sh = 16; sh; sh >>= 1) s += __shfl_xor(s, sh, 32);
    if (lane == 0) sm[wv] = s;
    __syncthreads();

    if (threadIdx.x == 0) {
        float tot = 0.0f;
#pragma unroll
        for (int w = 0; w < 8; ++w) tot += sm[w];
        const int t = targets[row];
        const float logp = lr[t] - m - __logf(tot);
        atomicAdd(accum, -logp * (1.0f / (float)N_BATCH));
    }
}

// ---------------- kernel 2a: fused bf16 hi/lo split + row squared norms -----
__global__ void convert_split_kernel(const float* __restrict__ feat,
                                     __bf16* __restrict__ f_hi,
                                     __bf16* __restrict__ f_lo,
                                     float* __restrict__ sq) {
    const int row = blockIdx.x;
    const float* fr = feat + (size_t)row * FEAT_D;
    __bf16* hr = f_hi + (size_t)row * FEAT_D;
    __bf16* lr = f_lo + (size_t)row * FEAT_D;
    __shared__ float sm[8];
    const int lane = threadIdx.x & 31;
    const int wv   = threadIdx.x >> 5;

    float s = 0.0f;
    for (int j = threadIdx.x; j < FEAT_D; j += 256) {
        float v = fr[j];
        BF16Pair p = split_bf16(v);
        hr[j] = p.hi;
        lr[j] = p.lo;
        s = fmaf(v, v, s);
    }
    for (int sh = 16; sh; sh >>= 1) s += __shfl_xor(s, sh, 32);
    if (lane == 0) sm[wv] = s;
    __syncthreads();
    if (threadIdx.x == 0) {
        float tot = 0.0f;
#pragma unroll
        for (int w = 0; w < 8; ++w) tot += sm[w];
        sq[row] = tot;
    }
}

// ---------------- kernel 2b: plain row squared norms (fallback path) --------
__global__ void sqnorm_kernel(const float* __restrict__ feat,
                              float* __restrict__ sq) {
    const int row = blockIdx.x;
    const float* fr = feat + (size_t)row * FEAT_D;
    __shared__ float sm[8];
    const int lane = threadIdx.x & 31;
    const int wv   = threadIdx.x >> 5;

    float s = 0.0f;
    for (int j = threadIdx.x; j < FEAT_D; j += 256) {
        float v = fr[j];
        s = fmaf(v, v, s);
    }
    for (int sh = 16; sh; sh >>= 1) s += __shfl_xor(s, sh, 32);
    if (lane == 0) sm[wv] = s;
    __syncthreads();
    if (threadIdx.x == 0) {
        float tot = 0.0f;
#pragma unroll
        for (int w = 0; w < 8; ++w) tot += sm[w];
        sq[row] = tot;
    }
}

// ---------------- kernel 3 (fast): symmetric Gram + mining ------------------
// 64x64 block tiles, 128 threads = 4 waves, each wave a 32x32 register tile
// (2x2 grid of 16x16 WMMA accumulators: 12 WMMAs per 8 fragment loads).
// Only bn >= bm computed (symmetry); off-diagonal tiles commit both the
// row-direction and column-direction mining results.
#define BTS 64
__global__ void gram_mine_sym_kernel(const __bf16* __restrict__ f_hi,
                                     const __bf16* __restrict__ f_lo,
                                     const float*  __restrict__ sq,
                                     unsigned* __restrict__ ap_bits,
                                     unsigned* __restrict__ an_bits) {
    const int bm = blockIdx.x;
    const int bn = blockIdx.y;
    if (bn < bm) return;                // lower-triangle blocks: skipped

    const int wave = threadIdx.x >> 5;
    const int lane = threadIdx.x & 31;
    const int half = lane >> 4;         // 0: lanes 0-15, 1: lanes 16-31
    const int l15  = lane & 15;

    const int m0 = bm * BTS + (wave & 1) * 32;   // wave tile: rows m0..m0+31
    const int n0 = bn * BTS + (wave >> 1) * 32;  //            cols n0..n0+31

    const size_t a0 = (size_t)(m0 + l15) * FEAT_D;        // rows m0..m0+15
    const size_t a1 = a0 + (size_t)16 * FEAT_D;           // rows m0+16..m0+31
    const size_t b0 = (size_t)(n0 + l15) * FEAT_D;        // cols n0..n0+15
    const size_t b1 = b0 + (size_t)16 * FEAT_D;           // cols n0+16..n0+31
    const int ka = half * 8;            // A: K halves {0..7,16..23} / {8..15,24..31}
    const int kb = half * 8;            // B: K halves {0..7,16..23} / {8..15,24..31}

    v8f acc[2][2] = {};

    for (int k = 0; k < FEAT_D; k += 128) {
        // prefetch next 128-elem K-block; each lane covers its own row's lines
        __builtin_prefetch(f_hi + a0 + k + 128, 0, 3);
        __builtin_prefetch(f_hi + a1 + k + 128, 0, 3);
        __builtin_prefetch(f_hi + b0 + k + 128, 0, 3);
        __builtin_prefetch(f_hi + b1 + k + 128, 0, 3);
        __builtin_prefetch(f_lo + a0 + k + 128, 0, 3);
        __builtin_prefetch(f_lo + a1 + k + 128, 0, 3);
        __builtin_prefetch(f_lo + b0 + k + 128, 0, 3);
        __builtin_prefetch(f_lo + b1 + k + 128, 0, 3);
#pragma unroll
        for (int kk = 0; kk < 128; kk += 32) {
            const int kc = k + kk;
            v16bf ah[2], al[2], bh[2], bl[2];
            ah[0] = load_frag(f_hi + a0 + kc + ka, 16);
            al[0] = load_frag(f_lo + a0 + kc + ka, 16);
            ah[1] = load_frag(f_hi + a1 + kc + ka, 16);
            al[1] = load_frag(f_lo + a1 + kc + ka, 16);
            bh[0] = load_frag(f_hi + b0 + kc + kb, 16);
            bl[0] = load_frag(f_lo + b0 + kc + kb, 16);
            bh[1] = load_frag(f_hi + b1 + kc + kb, 16);
            bl[1] = load_frag(f_lo + b1 + kc + kb, 16);
            // G ~= hi*hi + hi*lo + lo*hi (lo*lo negligible)
#pragma unroll
            for (int ti = 0; ti < 2; ++ti)
#pragma unroll
                for (int tj = 0; tj < 2; ++tj) {
                    acc[ti][tj] = __builtin_amdgcn_wmma_f32_16x16x32_bf16(
                        false, ah[ti], false, bh[tj], (short)0, acc[ti][tj], false, false);
                    acc[ti][tj] = __builtin_amdgcn_wmma_f32_16x16x32_bf16(
                        false, ah[ti], false, bl[tj], (short)0, acc[ti][tj], false, false);
                    acc[ti][tj] = __builtin_amdgcn_wmma_f32_16x16x32_bf16(
                        false, al[ti], false, bh[tj], (short)0, acc[ti][tj], false, false);
                }
        }
    }

    // ---- epilogue: distances + batch-hard mining (both directions) ----
    float sqn[2], colmin[2], colmax[2];
    int   ngrp[2];
#pragma unroll
    for (int tj = 0; tj < 2; ++tj) {
        const int n = n0 + tj * 16 + l15;
        sqn[tj]    = sq[n];
        ngrp[tj]   = n >> 2;            // identity group (NUM_INSTANCES = 4)
        colmin[tj] = BIG_DIST;
        colmax[tj] = 0.0f;
    }

#pragma unroll
    for (int ti = 0; ti < 2; ++ti) {
#pragma unroll
        for (int e = 0; e < 8; ++e) {
            const int m = m0 + ti * 16 + half * 8 + e;   // C layout: row per VGPR/half
            const float sqm  = sq[m];
            const int   mgrp = m >> 2;

            float vmin = BIG_DIST;
            float vmax = 0.0f;
#pragma unroll
            for (int tj = 0; tj < 2; ++tj) {
                const float d2   = sqm + sqn[tj] - 2.0f * acc[ti][tj][e];
                const float dist = sqrtf(fmaxf(d2, 1e-12f));
                const bool  same = (mgrp == ngrp[tj]);

                const float vm = same ? BIG_DIST : dist;
                const float vx = same ? dist     : 0.0f;
                colmin[tj] = fminf(colmin[tj], vm);
                colmax[tj] = fmaxf(colmax[tj], vx);
                vmin = fminf(vmin, vm);
                vmax = fmaxf(vmax, vx);
            }
            // row-direction: reduce across the 16 lanes holding row m
#pragma unroll
            for (int s = 8; s; s >>= 1) {
                vmin = fminf(vmin, __shfl_xor(vmin, s, 32));
                vmax = fmaxf(vmax, __shfl_xor(vmax, s, 32));
            }
            if (l15 == 0) {
                atomicMin((int*)&an_bits[m], __float_as_int(vmin));
                atomicMax((int*)&ap_bits[m], __float_as_int(vmax));
            }
        }
    }

    if (bn != bm) {   // transpose commit only off-diagonal (no double count)
#pragma unroll
        for (int tj = 0; tj < 2; ++tj) {
            const float cmin = fminf(colmin[tj], __shfl_xor(colmin[tj], 16, 32));
            const float cmax = fmaxf(colmax[tj], __shfl_xor(colmax[tj], 16, 32));
            if (half == 0) {
                const int n = n0 + tj * 16 + l15;
                atomicMin((int*)&an_bits[n], __float_as_int(cmin));
                atomicMax((int*)&ap_bits[n], __float_as_int(cmax));
            }
        }
    }
}

// ---------------- kernel 3 (fallback): fused-split Gram + mining ------------
#define TM 64
#define TN 32
__global__ void gram_mine_kernel(const float* __restrict__ feat,
                                 const float* __restrict__ sq,
                                 unsigned* __restrict__ ap_bits,
                                 unsigned* __restrict__ an_bits) {
    const int wave = threadIdx.x >> 5;
    const int lane = threadIdx.x & 31;
    const int half = lane >> 4;
    const int l15  = lane & 15;

    const int m0 = blockIdx.x * TM + (wave & 3) * 16;
    const int n0 = blockIdx.y * TN + (wave >> 2) * 16;

    const float* arow = feat + (size_t)(m0 + l15) * FEAT_D;
    const float* brow = feat + (size_t)(n0 + l15) * FEAT_D;
    const int ka = half * 16;
    const int kb = half * 16;

    v8f acc = {};
    for (int k = 0; k < FEAT_D; k += 32) {
        v16bf a_hi, a_lo, b_hi, b_lo;
#pragma unroll
        for (int e = 0; e < 8; ++e) {
            BF16Pair p0 = split_bf16(arow[k + ka + e]);
            a_hi[e]     = p0.hi;  a_lo[e]     = p0.lo;
            BF16Pair p1 = split_bf16(arow[k + ka + 8 + e]);
            a_hi[8 + e] = p1.hi;  a_lo[8 + e] = p1.lo;
        }
#pragma unroll
        for (int e = 0; e < 16; ++e) {
            BF16Pair p = split_bf16(brow[k + kb + e]);
            b_hi[e] = p.hi;  b_lo[e] = p.lo;
        }
        acc = __builtin_amdgcn_wmma_f32_16x16x32_bf16(
                  false, a_hi, false, b_hi, (short)0, acc, false, false);
        acc = __builtin_amdgcn_wmma_f32_16x16x32_bf16(
                  false, a_hi, false, b_lo, (short)0, acc, false, false);
        acc = __builtin_amdgcn_wmma_f32_16x16x32_bf16(
                  false, a_lo, false, b_hi, (short)0, acc, false, false);
    }

    const int   n    = n0 + l15;
    const float sqn  = sq[n];
    const int   ngrp = n >> 2;

#pragma unroll
    for (int e = 0; e < 8; ++e) {
        const int m = m0 + half * 8 + e;
        const float d2   = sq[m] + sqn - 2.0f * acc[e];
        const float dist = sqrtf(fmaxf(d2, 1e-12f));
        const bool  same = ((m >> 2) == ngrp);

        float vmin = same ? BIG_DIST : dist;
        float vmax = same ? dist     : 0.0f;
#pragma unroll
        for (int s = 8; s; s >>= 1) {
            vmin = fminf(vmin, __shfl_xor(vmin, s, 32));
            vmax = fmaxf(vmax, __shfl_xor(vmax, s, 32));
        }
        if (l15 == 0) {
            atomicMin((int*)&an_bits[m], __float_as_int(vmin));
            atomicMax((int*)&ap_bits[m], __float_as_int(vmax));
        }
    }
}

// ---------------- kernel 4: final combine ----------------
__global__ void final_kernel(const unsigned* __restrict__ ap_bits,
                             const unsigned* __restrict__ an_bits,
                             const float* __restrict__ accum,
                             float* __restrict__ out) {
    __shared__ float red[32];
    float s = 0.0f;
    for (int i = threadIdx.x; i < N_BATCH; i += 1024) {
        const float d_ap = __uint_as_float(ap_bits[i]);
        const float d_an = __uint_as_float(an_bits[i]);
        s += fmaxf(d_ap - d_an + MARGIN1, 0.0f);
    }
    for (int sh = 16; sh; sh >>= 1) s += __shfl_xor(s, sh, 32);
    if ((threadIdx.x & 31) == 0) red[threadIdx.x >> 5] = s;
    __syncthreads();
    if (threadIdx.x < 32) {
        float v = red[threadIdx.x];
        for (int sh = 16; sh; sh >>= 1) v += __shfl_xor(v, sh, 32);
        if (threadIdx.x == 0)
            out[0] = accum[0] + v * (1.0f / (float)N_BATCH);
    }
}

// ---------------- launcher ----------------
extern "C" void kernel_launch(void* const* d_in, const int* in_sizes, int n_in,
                              void* d_out, int out_size, void* d_ws, size_t ws_size,
                              hipStream_t stream) {
    const float* logits  = (const float*)d_in[0];   // [4096, 751]
    const float* feat    = (const float*)d_in[1];   // [4096, 2048]
    const int*   targets = (const int*)d_in[2];     // [4096]
    float* out = (float*)d_out;

    const size_t FEAT_ELEMS = (size_t)N_BATCH * FEAT_D;
    const size_t FEAT_BYTES = FEAT_ELEMS * sizeof(__bf16);   // 16 MB
    const size_t SMALL_BYTES = (size_t)N_BATCH * 12 + 256;   // sq+ap+an+acc
    const size_t BIG_BYTES   = 2 * FEAT_BYTES + SMALL_BYTES;

    char* base = (char*)d_ws;

    if (ws_size >= BIG_BYTES) {
        // ---- fast path: pre-split bf16 hi/lo + symmetric Gram ----
        __bf16*   f_hi   = (__bf16*)base;                          // 16 MB
        __bf16*   f_lo   = (__bf16*)(base + FEAT_BYTES);           // 16 MB
        float*    ws_sq  = (float*)(base + 2 * FEAT_BYTES);        // [4096]
        unsigned* ws_ap  = (unsigned*)(ws_sq + N_BATCH);           // [4096]
        unsigned* ws_an  = ws_ap + N_BATCH;                        // [4096]
        float*    ws_acc = (float*)(ws_an + N_BATCH);              // [1]

        init_kernel<<<(N_BATCH + 255) / 256, 256, 0, stream>>>(ws_ap, ws_an, ws_acc);
        xent_kernel<<<N_BATCH, 256, 0, stream>>>(logits, targets, ws_acc);
        convert_split_kernel<<<N_BATCH, 256, 0, stream>>>(feat, f_hi, f_lo, ws_sq);
        dim3 grid(N_BATCH / BTS, N_BATCH / BTS);
        gram_mine_sym_kernel<<<grid, 128, 0, stream>>>(f_hi, f_lo, ws_sq, ws_ap, ws_an);
        final_kernel<<<1, 1024, 0, stream>>>(ws_ap, ws_an, ws_acc, out);
    } else {
        // ---- fallback: on-the-fly split, full grid ----
        float*    ws_sq  = (float*)base;
        unsigned* ws_ap  = (unsigned*)(ws_sq + N_BATCH);
        unsigned* ws_an  = ws_ap + N_BATCH;
        float*    ws_acc = (float*)(ws_an + N_BATCH);

        init_kernel<<<(N_BATCH + 255) / 256, 256, 0, stream>>>(ws_ap, ws_an, ws_acc);
        xent_kernel<<<N_BATCH, 256, 0, stream>>>(logits, targets, ws_acc);
        sqnorm_kernel<<<N_BATCH, 256, 0, stream>>>(feat, ws_sq);
        dim3 grid(N_BATCH / TM, N_BATCH / TN);
        gram_mine_kernel<<<grid, 256, 0, stream>>>(feat, ws_sq, ws_ap, ws_an);
        final_kernel<<<1, 1024, 0, stream>>>(ws_ap, ws_an, ws_acc, out);
    }
}
